// DGCNN_59682865545780
// MI455X (gfx1250) — matrix operand order repaired and checked by reading, not compiled, not run on hardware
//
#include <hip/hip_runtime.h>
#include <hip/hip_bf16.h>

typedef __attribute__((ext_vector_type(16))) _Float16 v16h;
typedef __attribute__((ext_vector_type(8)))  _Float16 v8h;
typedef __attribute__((ext_vector_type(8)))  float    v8f;

#define NPTS  8192
#define NB    2
#define KNNK  20
#define NCELL 32768
#define LRELU 0.2f

// ---------------- helpers ----------------

__device__ __forceinline__ unsigned mapf(float x) {
  unsigned u = __float_as_uint(x);
  return (u & 0x80000000u) ? ~u : (u | 0x80000000u);
}
__device__ __forceinline__ float unmapf(unsigned u) {
  return (u & 0x80000000u) ? __uint_as_float(u & 0x7FFFFFFFu)
                           : __uint_as_float(~u);
}

// A-fragment (16x32 MxK, f16), row-major source (row pointer offset to k-chunk).
// lane's K set: K(e) = 16*(e>>3) + 8*hi + (e&7) -> two aligned 16B loads.
__device__ __forceinline__ v16h load_a(const _Float16* __restrict__ row, int hi) {
  const _Float16* p = row + 8 * hi;
  v8h lo = *(const v8h*)(p);
  v8h hh = *(const v8h*)(p + 16);
  return __builtin_shufflevector(lo, hh, 0, 1, 2, 3, 4, 5, 6, 7, 8, 9, 10, 11,
                                 12, 13, 14, 15);
}

// B-fragment (32x16 KxN, f16) from LDS laid out column-major: col points at the
// 32 contiguous K-values of column n. lane's K set: K(e) = 16*(e>>3) + 8*hi + (e&7).
__device__ __forceinline__ v16h load_b_col(const _Float16* __restrict__ col,
                                           int hi) {
  const _Float16* p = col + 8 * hi;
  v8h lo = *(const v8h*)(p);
  v8h hh = *(const v8h*)(p + 16);
  return __builtin_shufflevector(lo, hh, 0, 1, 2, 3, 4, 5, 6, 7, 8, 9, 10, 11,
                                 12, 13, 14, 15);
}

__device__ __forceinline__ v8f wmma16(v16h a, v16h b, v8f c) {
  return __builtin_amdgcn_wmma_f32_16x16x32_f16(false, a, false, b, (short)0, c,
                                                false, false);
}

// ---------------- KNN: fused Gram tile (WMMA) + top-20 ----------------
// grid (NPTS/16, NB), 32 threads. feat: [B,C,NPTS] f32.
__global__ __launch_bounds__(32) void knn_kernel(const float* __restrict__ feat,
                                                 int C, int Cpad,
                                                 int* __restrict__ knn_out) {
  __shared__ _Float16 Qh[16 * 64];   // queries, row-major [16][Cpad] (A matrix)
  __shared__ _Float16 Chh[16 * 72];  // candidates, col-major [n][Cpad+8]
  __shared__ float qq[16], cc[16];
  __shared__ float negd[16 * 16];
  __shared__ float MV[16][2][KNNK];
  __shared__ int   MI[16][2][KNNK];

  const int lane = threadIdx.x;
  const int m16 = lane & 15;
  const int hi  = lane >> 4;
  const int b = blockIdx.y;
  const int qbase = blockIdx.x * 16;
  const int cstr = Cpad + 8;  // column stride (halfs), mult of 8 -> 16B aligned

  for (int t = lane; t < 16 * Cpad; t += 32) {
    int m = t / Cpad, k = t - m * Cpad;
    Qh[t] = (k < C) ? (_Float16)feat[((size_t)(b * C + k)) * NPTS + qbase + m]
                    : (_Float16)0.0f;
  }
  if (lane < 16) {
    float s = 0.0f;
    for (int c = 0; c < C; ++c) {
      float v = feat[((size_t)(b * C + c)) * NPTS + qbase + lane];
      s += v * v;
    }
    qq[lane] = s;
  }

  float tv[KNNK]; int ti[KNNK];
#pragma unroll
  for (int p = 0; p < KNNK; ++p) { tv[p] = -3.4e38f; ti[p] = 0; }
  __syncthreads();

  for (int tile = 0; tile < NPTS / 16; ++tile) {
    const int cbase = tile * 16;
    __syncthreads();
    for (int t = lane; t < Cpad * 16; t += 32) {
      int k = t >> 4, n = t & 15;
      Chh[n * cstr + k] =
          (k < C) ? (_Float16)feat[((size_t)(b * C + k)) * NPTS + cbase + n]
                  : (_Float16)0.0f;
    }
    if (lane < 16) {
      float s = 0.0f;
      for (int c = 0; c < C; ++c) {
        float v = feat[((size_t)(b * C + c)) * NPTS + cbase + lane];
        s += v * v;
      }
      cc[lane] = s;
    }
    __syncthreads();

    v8f acc = {};
    for (int kc = 0; kc < Cpad; kc += 32) {
      v16h a  = load_a(Qh + m16 * Cpad + kc, hi);
      v16h bf = load_b_col(Chh + m16 * cstr + kc, hi);
      acc = wmma16(a, bf, acc);
    }
#pragma unroll
    for (int r = 0; r < 8; ++r) {
      int M = r + 8 * hi;  // query row
      negd[M * 16 + m16] = 2.0f * acc[r] - qq[M] - cc[m16];
    }
    __syncthreads();
    // lane owns query m16, candidate half [8*hi, 8*hi+8)
#pragma unroll
    for (int j = 0; j < 8; ++j) {
      int n = 8 * hi + j;
      float v = negd[m16 * 16 + n];
      if (v > tv[KNNK - 1]) {
        float cv = v; int ci = cbase + n;
#pragma unroll
        for (int p = 0; p < KNNK; ++p) {
          if (cv > tv[p]) {
            float tf = tv[p]; tv[p] = cv; cv = tf;
            int tj = ti[p]; ti[p] = ci; ci = tj;
          }
        }
      }
    }
  }
#pragma unroll
  for (int s = 0; s < KNNK; ++s) { MV[m16][hi][s] = tv[s]; MI[m16][hi][s] = ti[s]; }
  __syncthreads();
  if (lane < 16) {
    int p0 = 0, p1 = 0;
    size_t ob = ((size_t)b * NPTS + qbase + lane) * KNNK;
    for (int s = 0; s < KNNK; ++s) {
      float v0 = MV[lane][0][p0], v1 = MV[lane][1][p1];
      int i0 = MI[lane][0][p0], i1 = MI[lane][1][p1];
      bool t0 = (v0 > v1) || (v0 == v1 && i0 < i1);
      knn_out[ob + s] = t0 ? i0 : i1;
      if (t0) ++p0; else ++p1;
    }
  }
}

// ---------------- fused edge-conv block ----------------
// grid (NPTS/4, NB), 32 threads: 4 points x 20 edges = 5 column tiles.
// layerA: [64 x CA] on edge features (CA = padded 2C); optional layerB: [64 x 64];
// BN + LeakyReLU after each; max over K edges -> outf [B,64,NPTS].
__global__ __launch_bounds__(32) void edge_conv_kernel(
    const float* __restrict__ feat, int C, int CA,
    const int* __restrict__ knn,
    const _Float16* __restrict__ WAh, const float* __restrict__ gA,
    const float* __restrict__ bA,
    const _Float16* __restrict__ WBh, const float* __restrict__ gB,
    const float* __restrict__ bB, int hasB, float* __restrict__ outf) {
  __shared__ _Float16 Eh[16 * 136];  // edge tile, col-major [n][CA+8]
  __shared__ _Float16 Th[16 * 72];   // layer-A output, col-major [n][72]
  __shared__ float ctr[4 * 64];
  __shared__ int nbr[80];
  __shared__ unsigned int umax[64 * 4];

  const int lane = threadIdx.x;
  const int m16 = lane & 15, hi = lane >> 4;
  const int b = blockIdx.y;
  const int pbase = blockIdx.x * 4;
  const int estr = CA + 8;
  const float bnn = rsqrtf(1.0f + 1e-5f);

  for (int t = lane; t < 80; t += 32)
    nbr[t] = knn[((size_t)b * NPTS + pbase + t / KNNK) * KNNK + t % KNNK];
  for (int t = lane; t < 4 * C; t += 32) {
    int p = t / C, c = t - p * C;
    ctr[p * 64 + c] = feat[((size_t)(b * C + c)) * NPTS + pbase + p];
  }
  for (int t = lane; t < 256; t += 32) umax[t] = 0u;
  __syncthreads();

  for (int tile = 0; tile < 5; ++tile) {
    const int ebase = tile * 16;
    __syncthreads();
    for (int t = lane; t < CA * 16; t += 32) {
      int k = t >> 4, n = t & 15;
      int e = ebase + n;
      int p = e / KNNK;
      float val = 0.0f;
      if (k < C) {
        int j = nbr[e];
        val = feat[((size_t)(b * C + k)) * NPTS + j] - ctr[p * 64 + k];
      } else if (k < 2 * C) {
        val = ctr[p * 64 + (k - C)];
      }
      Eh[n * estr + k] = (_Float16)val;
    }
    __syncthreads();

    const int ecol = ebase + m16;
    const int pcol = ecol / KNNK;

#pragma unroll
    for (int mt = 0; mt < 4; ++mt) {
      v8f acc = {};
      for (int kc = 0; kc < CA; kc += 32) {
        v16h a  = load_a(WAh + (size_t)(mt * 16 + m16) * CA + kc, hi);
        v16h bf = load_b_col(Eh + m16 * estr + kc, hi);
        acc = wmma16(a, bf, acc);
      }
#pragma unroll
      for (int r = 0; r < 8; ++r) {
        int o = mt * 16 + r + 8 * hi;
        float y = acc[r] * (gA[o] * bnn) + bA[o];
        y = (y >= 0.0f) ? y : LRELU * y;
        if (hasB) Th[m16 * 72 + o] = (_Float16)y;
        else atomicMax(&umax[o * 4 + pcol], mapf(y));
      }
    }
    if (hasB) {
      __syncthreads();
#pragma unroll
      for (int mt = 0; mt < 4; ++mt) {
        v8f acc = {};
#pragma unroll
        for (int kc = 0; kc < 64; kc += 32) {
          v16h a  = load_a(WBh + (size_t)(mt * 16 + m16) * 64 + kc, hi);
          v16h bf = load_b_col(Th + m16 * 72 + kc, hi);
          acc = wmma16(a, bf, acc);
        }
#pragma unroll
        for (int r = 0; r < 8; ++r) {
          int o = mt * 16 + r + 8 * hi;
          float y = acc[r] * (gB[o] * bnn) + bB[o];
          y = (y >= 0.0f) ? y : LRELU * y;
          atomicMax(&umax[o * 4 + pcol], mapf(y));
        }
      }
    }
  }
  __syncthreads();
  for (int t = lane; t < 256; t += 32) {
    int o = t >> 2, p = t & 3;
    outf[((size_t)(b * 64 + o)) * NPTS + pbase + p] = unmapf(umax[o * 4 + p]);
  }
}

// ---------------- pointwise MLP GEMM (up to 4 concatenated sources) -------
// grid (NPTS/16, O/64, NB), 32 threads. Wave computes 64x16 output tile.
__global__ __launch_bounds__(32) void mlp_kernel(
    const float* __restrict__ s0, int c0, int bc0,
    const float* __restrict__ s1, int c1,
    const float* __restrict__ s2, int c2,
    const float* __restrict__ s3, int c3,
    int Ctot, const _Float16* __restrict__ Wh,
    const float* __restrict__ g, const float* __restrict__ bb, int O,
    float* __restrict__ outf, unsigned int* __restrict__ gmax) {
  __shared__ _Float16 Xh[16 * 40];  // chunk, col-major [n][40]
  const int lane = threadIdx.x;
  const int m16 = lane & 15, hi = lane >> 4;
  const int b = blockIdx.z;
  const int nb = blockIdx.x * 16;
  const int mb = blockIdx.y * 64;
  const float bnn = rsqrtf(1.0f + 1e-5f);

  v8f acc[4] = {};
  for (int kc = 0; kc < Ctot; kc += 32) {
    __syncthreads();
    for (int t = lane; t < 32 * 16; t += 32) {
      int k = kc + (t >> 4), n = t & 15;
      float val;
      int c = k;
      if (c < c0) {
        val = bc0 ? s0[(size_t)b * c0 + c]
                  : s0[((size_t)(b * c0 + c)) * NPTS + nb + n];
      } else if ((c -= c0) < c1) {
        val = s1[((size_t)(b * c1 + c)) * NPTS + nb + n];
      } else if ((c -= c1) < c2) {
        val = s2[((size_t)(b * c2 + c)) * NPTS + nb + n];
      } else {
        c -= c2;
        val = s3[((size_t)(b * c3 + c)) * NPTS + nb + n];
      }
      Xh[n * 40 + (t >> 4)] = (_Float16)val;
    }
    __syncthreads();
    v16h bf = load_b_col(Xh + m16 * 40, hi);
#pragma unroll
    for (int mt = 0; mt < 4; ++mt) {
      v16h a = load_a(Wh + (size_t)(mb + mt * 16 + m16) * Ctot + kc, hi);
      acc[mt] = wmma16(a, bf, acc[mt]);
    }
  }
#pragma unroll
  for (int mt = 0; mt < 4; ++mt) {
#pragma unroll
    for (int r = 0; r < 8; ++r) {
      int o = mb + mt * 16 + r + 8 * hi;
      float y = acc[mt][r] * (g[o] * bnn) + bb[o];
      y = (y >= 0.0f) ? y : LRELU * y;
      if (outf) outf[((size_t)(b * O + o)) * NPTS + nb + m16] = y;
      if (gmax) {
        float m = y;
#pragma unroll
        for (int sft = 8; sft >= 1; sft >>= 1)
          m = fmaxf(m, __shfl_xor(m, sft, 16));
        if (m16 == 0) atomicMax(&gmax[b * O + o], mapf(m));
      }
    }
  }
}

// ---------------- small utility kernels ----------------

__global__ void zero_u32_kernel(unsigned int* __restrict__ p, int n) {
  int i = blockIdx.x * blockDim.x + threadIdx.x;
  if (i < n) p[i] = 0u;
}

__global__ void transpose_pts_kernel(const float* __restrict__ pts,
                                     float* __restrict__ ptst) {
  int i = blockIdx.x * blockDim.x + threadIdx.x;
  if (i < NB * NPTS * 3) {
    int b = i / (NPTS * 3);
    int r = i - b * NPTS * 3;
    int n = r / 3, c = r - n * 3;
    ptst[((size_t)(b * 3 + c)) * NPTS + n] = pts[i];
  }
}

__global__ void convert_w_kernel(const float* __restrict__ W,
                                 _Float16* __restrict__ Wh, int O, int Cin,
                                 int Cpad) {
  int i = blockIdx.x * blockDim.x + threadIdx.x;
  if (i < O * Cpad) {
    int o = i / Cpad, c = i - o * Cpad;
    Wh[i] = (c < Cin) ? (_Float16)W[o * Cin + c] : (_Float16)0.0f;
  }
}

__global__ void unmap_kernel(const unsigned int* __restrict__ u,
                             float* __restrict__ f, int n) {
  int i = blockIdx.x * blockDim.x + threadIdx.x;
  if (i < n) f[i] = unmapf(u[i]);
}

__global__ void cell_kernel(const float* __restrict__ pts,
                            int* __restrict__ cells, float* __restrict__ cnt) {
  int i = blockIdx.x * blockDim.x + threadIdx.x;
  if (i < NB * NPTS) {
    int b = i / NPTS;
    int cd[3];
#pragma unroll
    for (int d = 0; d < 3; ++d) {
      float p = pts[(size_t)i * 3 + d] / 1.101f + 0.5f;
      p = fminf(fmaxf(p, 0.0f), 0.999f);
      cd[d] = (int)(p * 32.0f);
    }
    int cell = cd[0] + 32 * (cd[1] + 32 * cd[2]);
    cells[i] = cell;
    atomicAdd(&cnt[b * NCELL + cell], 1.0f);
  }
}

__global__ void scatter_kernel(const float* __restrict__ x7,
                               const int* __restrict__ cells,
                               float* __restrict__ outp) {
  int bn = blockIdx.x;
  int ch = threadIdx.x;
  int b = bn / NPTS, n = bn - b * NPTS;
  int cell = cells[bn];
  float v = x7[((size_t)(b * 128 + ch)) * NPTS + n];
  atomicAdd(&outp[((size_t)(b * 128 + ch)) * NCELL + cell], v);
}

__global__ void normalize_kernel(float* __restrict__ outp,
                                 const float* __restrict__ cnt) {
  int i = blockIdx.x * blockDim.x + threadIdx.x;
  if (i < NB * 128 * NCELL) {
    int b = i / (128 * NCELL);
    int cell = i % NCELL;
    outp[i] = outp[i] / fmaxf(cnt[b * NCELL + cell], 1.0f);
  }
}

// ---------------- host ----------------

extern "C" void kernel_launch(void* const* d_in, const int* in_sizes, int n_in,
                              void* d_out, int out_size, void* d_ws,
                              size_t ws_size, hipStream_t stream) {
  (void)in_sizes; (void)n_in; (void)out_size; (void)ws_size;
  const float* pts = (const float*)d_in[0];
  const float* W[8]; const float* gg[8]; const float* bv[8];
  for (int i = 0; i < 8; ++i) {
    W[i]  = (const float*)d_in[1 + 3 * i];
    gg[i] = (const float*)d_in[2 + 3 * i];
    bv[i] = (const float*)d_in[3 + 3 * i];
  }

  char* wp = (char*)d_ws;
  auto alloc = [&](size_t bytes) {
    void* p = (void*)wp;
    wp += (bytes + 255) & ~(size_t)255;
    return p;
  };
  float* ptst = (float*)alloc((size_t)NB * 3 * NPTS * 4);
  int* idxb   = (int*)alloc((size_t)NB * NPTS * KNNK * 4);
  float* f1   = (float*)alloc((size_t)NB * 64 * NPTS * 4);
  float* f2   = (float*)alloc((size_t)NB * 64 * NPTS * 4);
  float* f3   = (float*)alloc((size_t)NB * 64 * NPTS * 4);
  float* x6   = (float*)alloc((size_t)NB * 512 * NPTS * 4);
  float* x7   = (float*)alloc((size_t)NB * 128 * NPTS * 4);
  unsigned* globu = (unsigned*)alloc((size_t)NB * 1024 * 4);
  float* globf    = (float*)alloc((size_t)NB * 1024 * 4);
  float* cnt  = (float*)alloc((size_t)NB * NCELL * 4);
  int* cells  = (int*)alloc((size_t)NB * NPTS * 4);

  static const int Od[8] = {64, 64, 64, 64, 64, 1024, 512, 128};
  static const int Cd[8] = {6, 64, 128, 64, 128, 192, 1216, 512};
  static const int Cp[8] = {32, 64, 128, 64, 128, 192, 1216, 512};
  _Float16* Wh[8];
  for (int i = 0; i < 8; ++i) Wh[i] = (_Float16*)alloc((size_t)Od[i] * Cp[i] * 2);

  transpose_pts_kernel<<<(NB * NPTS * 3 + 255) / 256, 256, 0, stream>>>(pts, ptst);
  for (int i = 0; i < 8; ++i)
    convert_w_kernel<<<(Od[i] * Cp[i] + 255) / 256, 256, 0, stream>>>(
        W[i], Wh[i], Od[i], Cd[i], Cp[i]);

  dim3 kg(NPTS / 16, NB);
  dim3 eg(NPTS / 4, NB);

  // EdgeConv block 1 (points, C=3)
  knn_kernel<<<kg, 32, 0, stream>>>(ptst, 3, 32, idxb);
  edge_conv_kernel<<<eg, 32, 0, stream>>>(ptst, 3, 32, idxb, Wh[0], gg[0], bv[0],
                                          Wh[1], gg[1], bv[1], 1, f1);
  // EdgeConv block 2 (f1, C=64)
  knn_kernel<<<kg, 32, 0, stream>>>(f1, 64, 64, idxb);
  edge_conv_kernel<<<eg, 32, 0, stream>>>(f1, 64, 128, idxb, Wh[2], gg[2], bv[2],
                                          Wh[3], gg[3], bv[3], 1, f2);
  // EdgeConv block 3 (f2, C=64), single layer
  knn_kernel<<<kg, 32, 0, stream>>>(f2, 64, 64, idxb);
  edge_conv_kernel<<<eg, 32, 0, stream>>>(f2, 64, 128, idxb, Wh[4], gg[4], bv[4],
                                          Wh[4], gg[4], bv[4], 0, f3);

  // Layer 5: 192 -> 1024, fused global max (x5 never materialized)
  zero_u32_kernel<<<(NB * 1024 + 255) / 256, 256, 0, stream>>>(globu, NB * 1024);
  mlp_kernel<<<dim3(NPTS / 16, 1024 / 64, NB), 32, 0, stream>>>(
      f1, 64, 0, f2, 64, f3, 64, f3, 0, 192, Wh[5], gg[5], bv[5], 1024, nullptr,
      globu);
  unmap_kernel<<<(NB * 1024 + 255) / 256, 256, 0, stream>>>(globu, globf,
                                                            NB * 1024);
  // Layer 6: concat(glob,f1,f2,f3) 1216 -> 512
  mlp_kernel<<<dim3(NPTS / 16, 512 / 64, NB), 32, 0, stream>>>(
      globf, 1024, 1, f1, 64, f2, 64, f3, 64, 1216, Wh[6], gg[6], bv[6], 512, x6,
      nullptr);
  // Layer 7: 512 -> 128
  mlp_kernel<<<dim3(NPTS / 16, 128 / 64, NB), 32, 0, stream>>>(
      x6, 512, 0, x6, 0, x6, 0, x6, 0, 512, Wh[7], gg[7], bv[7], 128, x7,
      nullptr);

  // Grid features: scatter-mean into [B,128,32^3]
  zero_u32_kernel<<<(NB * NCELL + 255) / 256, 256, 0, stream>>>((unsigned*)cnt,
                                                                NB * NCELL);
  zero_u32_kernel<<<(NB * 128 * NCELL + 255) / 256, 256, 0, stream>>>(
      (unsigned*)d_out, NB * 128 * NCELL);
  cell_kernel<<<(NB * NPTS + 255) / 256, 256, 0, stream>>>(pts, cells, cnt);
  scatter_kernel<<<NB * NPTS, 128, 0, stream>>>(x7, cells, (float*)d_out);
  normalize_kernel<<<(NB * 128 * NCELL + 255) / 256, 256, 0, stream>>>(
      (float*)d_out, cnt);
}
